// GATNet_63866163692310
// MI455X (gfx1250) — compile-verified, matrix operand order checked by reading, not run on hardware
//
#include <hip/hip_runtime.h>
#include <hip/hip_bf16.h>

typedef __bf16 bf16;
typedef __bf16 v16bf __attribute__((ext_vector_type(16)));
typedef __bf16 v8bf  __attribute__((ext_vector_type(8)));
typedef float  v8f   __attribute__((ext_vector_type(8)));

#define NNODES 8192
#define NEDGES 65536
#define ETOT   (NEDGES + NNODES)
#define NEG_SLOPE 0.2f

// ---------------------------------------------------------------- utilities
__global__ void fill_f32(float* __restrict__ p, float v, int n) {
  int i = blockIdx.x * blockDim.x + threadIdx.x;
  if (i < n) p[i] = v;
}

// fp32 -> bf16 row-major copy (activations; cols == ldout here)
__global__ void cvt_bf16(const float* __restrict__ in, bf16* __restrict__ out, int n) {
  int i = blockIdx.x * blockDim.x + threadIdx.x;
  if (i < n) out[i] = (bf16)in[i];
}

// fp32 W (K x cols, row-major) -> bf16 W^T (ldn x K, row-major), zero-padding
// columns cols..ldn-1.  LDS-tiled 32x32 transpose for coalescing.
__global__ __launch_bounds__(256)
void cvt_transpose_bf16(const float* __restrict__ in, bf16* __restrict__ out,
                        int K, int cols, int ldn) {
  __shared__ float t[32][33];
  int kb = blockIdx.x * 32, cb = blockIdx.y * 32;
  for (int j = threadIdx.y; j < 32; j += 8) {
    int c = cb + threadIdx.x;
    t[j][threadIdx.x] = (c < cols) ? in[(size_t)(kb + j) * cols + c] : 0.f;
  }
  __syncthreads();
  for (int j = threadIdx.y; j < 32; j += 8) {
    int c = cb + j, k = kb + threadIdx.x;
    out[(size_t)c * K + k] = (bf16)t[threadIdx.x][j];
  }
}

// ---------------------------------------------------------------- WMMA GEMM
// C(M x ldn, f32) = A(M x K, bf16 row-major) * BT(ldn x K, bf16 row-major)^T
// 128x128 tile / 256 threads (8 wave32s as 2x4), K-step 32, double-buffered
// LDS staged via GLOBAL_LOAD_ASYNC_TO_LDS_B128 (ASYNCcnt).
#define TM 128
#define TN 128
#define TK 32
#define LDT 40   // LDS row stride (elements): 80B, 16B-aligned rows

__device__ __forceinline__ void async_b128(const bf16* lds, const bf16* g) {
  // low 32 bits of a generic LDS pointer == segment-relative LDS byte address
  unsigned loff = (unsigned)(unsigned long long)lds;
  unsigned long long ga = (unsigned long long)g;
  asm volatile("global_load_async_to_lds_b128 %0, %1, off"
               :: "v"(loff), "v"(ga) : "memory");
}
__device__ __forceinline__ void wait_async0() {
  asm volatile("s_wait_asynccnt 0x0" ::: "memory");
}

__global__ __launch_bounds__(256)
void gemm_bf16_wmma(const bf16* __restrict__ A, const bf16* __restrict__ BT,
                    float* __restrict__ C, int K, int ldn) {
  __shared__ bf16 As[2][TM * LDT];
  __shared__ bf16 Bs[2][TN * LDT];

  const int tid  = threadIdx.x;
  const int lane = tid & 31;
  const int wave = tid >> 5;
  const int wm   = wave & 1;    // 0..1 -> 64-row slab
  const int wn   = wave >> 1;   // 0..3 -> 32-col slab
  const int row0 = blockIdx.x * TM;
  const int col0 = blockIdx.y * TN;

  // cooperative loader: thread t stages 16 contiguous bf16 (2 x b128) per tile
  const int srow = tid >> 1, scol = (tid & 1) * 16;
  const bf16* agp = A  + (size_t)(row0 + srow) * K + scol;
  const bf16* bgp = BT + (size_t)(col0 + srow) * K + scol;
  bf16* alp = &As[0][srow * LDT + scol];
  bf16* blp = &Bs[0][srow * LDT + scol];
  const int bufstride = TM * LDT;

  // fragment lane mapping (16-bit A/B layout: lanes 0-15 K0-7/16-23, 16-31 K8-15/24-31)
  const int lr = lane & 15;
  const int kh = lane >> 4;

  v8f acc[4][2];
#pragma unroll
  for (int m = 0; m < 4; ++m)
#pragma unroll
    for (int n = 0; n < 2; ++n)
#pragma unroll
      for (int j = 0; j < 8; ++j) acc[m][n][j] = 0.0f;

  // preload K-tile 0 into buffer 0
  async_b128(alp,     agp);
  async_b128(alp + 8, agp + 8);
  async_b128(blp,     bgp);
  async_b128(blp + 8, bgp + 8);

  int cur = 0;
  for (int k0 = 0; k0 < K; k0 += TK) {
    wait_async0();       // my slice of buffer `cur` landed
    __syncthreads();     // everyone's slice landed; prior reads of other buf done

    if (k0 + TK < K) {   // issue next K-tile into the other buffer
      int nb = cur ^ 1;
      const bf16* ag = agp + k0 + TK;
      const bf16* bg = bgp + k0 + TK;
      async_b128(alp + nb * bufstride,     ag);
      async_b128(alp + nb * bufstride + 8, ag + 8);
      async_b128(blp + nb * bufstride,     bg);
      async_b128(blp + nb * bufstride + 8, bg + 8);
    }

    v16bf bfrag[2];
#pragma unroll
    for (int n = 0; n < 2; ++n) {
      const bf16* bp = &Bs[cur][(wn * 32 + n * 16 + lr) * LDT];
      union { v16bf v; v8bf h[2]; } u;
      u.h[0] = *(const v8bf*)(bp + kh * 8);
      u.h[1] = *(const v8bf*)(bp + 16 + kh * 8);
      bfrag[n] = u.v;
    }
#pragma unroll
    for (int m = 0; m < 4; ++m) {
      const bf16* ap = &As[cur][(wm * 64 + m * 16 + lr) * LDT];
      union { v16bf v; v8bf h[2]; } u;
      u.h[0] = *(const v8bf*)(ap + kh * 8);
      u.h[1] = *(const v8bf*)(ap + 16 + kh * 8);
      v16bf afrag = u.v;
#pragma unroll
      for (int n = 0; n < 2; ++n) {
        acc[m][n] = __builtin_amdgcn_wmma_f32_16x16x32_bf16(
            false, afrag, false, bfrag[n], (short)0, acc[m][n], false, false);
      }
    }
    cur ^= 1;
  }

  // D layout: VGPR j -> row M=j (lanes 0-15) / M=j+8 (lanes 16-31), N = lane%16
#pragma unroll
  for (int m = 0; m < 4; ++m)
#pragma unroll
    for (int n = 0; n < 2; ++n) {
      int r0 = row0 + wm * 64 + m * 16 + kh * 8;
      int c  = col0 + wn * 32 + n * 16 + lr;
      float* cp = C + (size_t)r0 * ldn + c;
#pragma unroll
      for (int j = 0; j < 8; ++j) { *cp = acc[m][n][j]; cp += ldn; }
    }
}

// ---------------------------------------------------------------- attention
__global__ void attn_logits(const float* __restrict__ h,
                            const float* __restrict__ asrc,
                            const float* __restrict__ adst,
                            float* __restrict__ als, float* __restrict__ ald,
                            int H, int C, int ldh) {
  int nh = blockIdx.x;
  int n = nh / H, head = nh - n * H;
  const float* hp = h + (size_t)n * ldh + head * C;
  const float* as = asrc + head * C;
  const float* ad = adst + head * C;
  float s0 = 0.f, s1 = 0.f;
  for (int c = threadIdx.x; c < C; c += blockDim.x) {
    float v = hp[c];
    s0 += v * as[c];
    s1 += v * ad[c];
  }
  __shared__ float r0[128], r1[128];
  r0[threadIdx.x] = s0; r1[threadIdx.x] = s1;
  __syncthreads();
  for (int s = 64; s > 0; s >>= 1) {
    if (threadIdx.x < s) {
      r0[threadIdx.x] += r0[threadIdx.x + s];
      r1[threadIdx.x] += r1[threadIdx.x + s];
    }
    __syncthreads();
  }
  if (threadIdx.x == 0) { als[nh] = r0[0]; ald[nh] = r1[0]; }
}

__device__ __forceinline__ int edge_src(const int* ei, int e) {
  return (e < NEDGES) ? ei[e] : (e - NEDGES);
}
__device__ __forceinline__ int edge_dst(const int* ei, int e) {
  return (e < NEDGES) ? ei[NEDGES + e] : (e - NEDGES);
}

// float atomic max via signed/unsigned integer ordering trick
__device__ __forceinline__ void atomicMaxF(float* addr, float val) {
  if (val >= 0.f) atomicMax((int*)addr, __float_as_int(val));
  else            atomicMin((unsigned int*)addr, __float_as_uint(val));
}

__global__ void edge_logit_max(const int* __restrict__ ei,
                               const float* __restrict__ als,
                               const float* __restrict__ ald,
                               float* __restrict__ ebuf, float* __restrict__ m, int H) {
  int i = blockIdx.x * blockDim.x + threadIdx.x;
  if (i >= ETOT * H) return;
  int e = i / H, head = i - e * H;
  int s = edge_src(ei, e), d = edge_dst(ei, e);
  float x = als[s * H + head] + ald[d * H + head];
  x = (x > 0.f) ? x : NEG_SLOPE * x;                 // leaky_relu
  ebuf[i] = x;
  atomicMaxF(&m[d * H + head], x);
}

__global__ void edge_exp_sum(const int* __restrict__ ei, float* __restrict__ ebuf,
                             const float* __restrict__ m, float* __restrict__ z, int H) {
  int i = blockIdx.x * blockDim.x + threadIdx.x;
  if (i >= ETOT * H) return;
  int e = i / H, head = i - e * H;
  int d = edge_dst(ei, e);
  float ex = __expf(ebuf[i] - m[d * H + head]);
  ebuf[i] = ex;
  atomicAdd(&z[d * H + head], ex);
}

__global__ void edge_alpha(const int* __restrict__ ei, float* __restrict__ ebuf,
                           const float* __restrict__ z, int H) {
  int i = blockIdx.x * blockDim.x + threadIdx.x;
  if (i >= ETOT * H) return;
  int e = i / H, head = i - e * H;
  int d = edge_dst(ei, e);
  ebuf[i] = ebuf[i] / (z[d * H + head] + 1e-16f);
}

// one block per edge: agg[dst] += h[src] * alpha
__global__ void edge_aggregate(const int* __restrict__ ei, const float* __restrict__ hbuf,
                               const float* __restrict__ alpha, float* __restrict__ agg,
                               int H, int C, int ldh) {
  int e = blockIdx.x;
  int s = edge_src(ei, e), d = edge_dst(ei, e);
  const float* hs = hbuf + (size_t)s * ldh;
  float* ag = agg + (size_t)d * ldh;
  int dtot = H * C;
  for (int idx = threadIdx.x; idx < dtot; idx += blockDim.x) {
    int head = idx / C;
    atomicAdd(&ag[idx], hs[idx] * alpha[e * H + head]);
  }
}

// relu -> layernorm -> bf16 (next layer's GEMM input)
__global__ void relu_ln_bf16(const float* __restrict__ agg, const float* __restrict__ bias,
                             const float* __restrict__ g, const float* __restrict__ beta,
                             bf16* __restrict__ out, int dtot, int ldh) {
  int r = blockIdx.x;
  const float* ap = agg + (size_t)r * ldh;
  float s = 0.f, ss = 0.f;
  for (int c = threadIdx.x; c < dtot; c += 256) {
    float v = fmaxf(ap[c] + bias[c], 0.f);
    s += v; ss += v * v;
  }
  __shared__ float rs[256], rss[256];
  rs[threadIdx.x] = s; rss[threadIdx.x] = ss;
  __syncthreads();
  for (int k = 128; k > 0; k >>= 1) {
    if (threadIdx.x < k) {
      rs[threadIdx.x]  += rs[threadIdx.x + k];
      rss[threadIdx.x] += rss[threadIdx.x + k];
    }
    __syncthreads();
  }
  float mean = rs[0] / (float)dtot;
  float var  = rss[0] / (float)dtot - mean * mean;
  float inv  = rsqrtf(var + 1e-5f);
  for (int c = threadIdx.x; c < dtot; c += 256) {
    float v = fmaxf(ap[c] + bias[c], 0.f);
    out[(size_t)r * dtot + c] = (bf16)((v - mean) * inv * g[c] + beta[c]);
  }
}

__global__ void add_bias_out(const float* __restrict__ agg, const float* __restrict__ b,
                             float* __restrict__ out, int C, int ldh, int n) {
  int i = blockIdx.x * blockDim.x + threadIdx.x;
  if (i >= n * C) return;
  int r = i / C, c = i - r * C;
  out[i] = agg[(size_t)r * ldh + c] + b[c];
}

// ---------------------------------------------------------------- driver
static inline size_t align256(size_t x) { return (x + 255) & ~(size_t)255; }

extern "C" void kernel_launch(void* const* d_in, const int* in_sizes, int n_in,
                              void* d_out, int out_size, void* d_ws, size_t ws_size,
                              hipStream_t stream) {
  (void)in_sizes; (void)n_in; (void)out_size; (void)ws_size;
  const float* x  = (const float*)d_in[0];
  const int*   ei = (const int*)d_in[1];
  const float* Wp[4]   = {(const float*)d_in[2],  (const float*)d_in[8],
                          (const float*)d_in[14], (const float*)d_in[20]};
  const float* Asrc[4] = {(const float*)d_in[3],  (const float*)d_in[9],
                          (const float*)d_in[15], (const float*)d_in[21]};
  const float* Adst[4] = {(const float*)d_in[4],  (const float*)d_in[10],
                          (const float*)d_in[16], (const float*)d_in[22]};
  const float* Bia[4]  = {(const float*)d_in[5],  (const float*)d_in[11],
                          (const float*)d_in[17], (const float*)d_in[23]};
  const float* Gam[3]  = {(const float*)d_in[6],  (const float*)d_in[12],
                          (const float*)d_in[18]};
  const float* Bet[3]  = {(const float*)d_in[7],  (const float*)d_in[13],
                          (const float*)d_in[19]};

  const int din[4]  = {256, 3584, 3072, 2048};
  const int Cd[4]   = {448, 384, 256, 1000};
  const int Hn[4]   = {8, 8, 8, 1};
  const int ldh[4]  = {3584, 3072, 2048, 1024};   // padded to multiple of 128

  // ---- workspace carve-up
  char* ws = (char*)d_ws;
  size_t off = 0;
  bf16* wbT[4];                                    // transposed bf16 weights: ldn x K
  for (int i = 0; i < 4; ++i) {
    wbT[i] = (bf16*)(ws + off);
    off = align256(off + (size_t)din[i] * ldh[i] * sizeof(bf16));
  }
  bf16*  xb   = (bf16*)(ws + off);  off = align256(off + (size_t)NNODES * 3584 * sizeof(bf16));
  float* h    = (float*)(ws + off); off = align256(off + (size_t)NNODES * 3584 * sizeof(float));
  float* agg  = (float*)(ws + off); off = align256(off + (size_t)NNODES * 3584 * sizeof(float));
  float* als  = (float*)(ws + off); off = align256(off + (size_t)NNODES * 8 * sizeof(float));
  float* ald  = (float*)(ws + off); off = align256(off + (size_t)NNODES * 8 * sizeof(float));
  float* mseg = (float*)(ws + off); off = align256(off + (size_t)NNODES * 8 * sizeof(float));
  float* zseg = (float*)(ws + off); off = align256(off + (size_t)NNODES * 8 * sizeof(float));
  float* ebuf = (float*)(ws + off); off = align256(off + (size_t)ETOT * 8 * sizeof(float));

  // ---- weights -> bf16, transposed to (ldn x K); layer-4 cols zero-padded to 1024
  for (int i = 0; i < 4; ++i) {
    dim3 g(din[i] / 32, ldh[i] / 32), b(32, 8);
    cvt_transpose_bf16<<<g, b, 0, stream>>>(Wp[i], wbT[i], din[i], Hn[i] * Cd[i], ldh[i]);
  }
  // x -> bf16 (row-major, K=256)
  {
    int tot = NNODES * 256;
    cvt_bf16<<<(tot + 255) / 256, 256, 0, stream>>>(x, xb, tot);
  }

  for (int L = 0; L < 4; ++L) {
    const int K = din[L], C = Cd[L], H = Hn[L], ld = ldh[L], dtot = H * C;

    dim3 grid(NNODES / TM, ld / TN);
    gemm_bf16_wmma<<<grid, 256, 0, stream>>>(xb, wbT[L], h, K, ld);

    attn_logits<<<NNODES * H, 128, 0, stream>>>(h, Asrc[L], Adst[L], als, ald, H, C, ld);

    fill_f32<<<(NNODES * H + 255) / 256, 256, 0, stream>>>(mseg, -3.402823466e38f, NNODES * H);
    fill_f32<<<(NNODES * H + 255) / 256, 256, 0, stream>>>(zseg, 0.f, NNODES * H);
    {
      int tot = NNODES * ld;
      fill_f32<<<(tot + 255) / 256, 256, 0, stream>>>(agg, 0.f, tot);
    }

    const int nEH = ETOT * H;
    edge_logit_max<<<(nEH + 255) / 256, 256, 0, stream>>>(ei, als, ald, ebuf, mseg, H);
    edge_exp_sum<<<(nEH + 255) / 256, 256, 0, stream>>>(ei, ebuf, mseg, zseg, H);
    edge_alpha<<<(nEH + 255) / 256, 256, 0, stream>>>(ei, ebuf, zseg, H);
    edge_aggregate<<<ETOT, 256, 0, stream>>>(ei, h, ebuf, agg, H, C, ld);

    if (L < 3) {
      relu_ln_bf16<<<NNODES, 256, 0, stream>>>(agg, Bia[L], Gam[L], Bet[L], xb, dtot, ld);
    } else {
      int tot = NNODES * 1000;
      add_bias_out<<<(tot + 255) / 256, 256, 0, stream>>>(agg, Bia[3], (float*)d_out,
                                                          1000, ld, NNODES);
    }
  }
}
